// LSTM_10118942949527
// MI455X (gfx1250) — matrix operand order repaired and checked by reading, not compiled, not run on hardware
//
#include <hip/hip_runtime.h>
#include <cmath>

#define SEQ   512
#define BATCH 64
#define DLAT  1024
#define DCAT  2048   // DLAT + DIN

typedef __bf16 bf16;
typedef __bf16 v16bf __attribute__((ext_vector_type(16)));
typedef float  v8f   __attribute__((ext_vector_type(8)));
typedef unsigned int u32x4 __attribute__((ext_vector_type(4)));
typedef float  f32x4 __attribute__((ext_vector_type(4)));

union Frag16 { v16bf v; u32x4 q[2]; };

// ---------------------------------------------------------------------------
// Pack W_all (f,i,o,c stacked -> 4096 x 2048) into WMMA B-fragment order:
//   fragment (n16, ks) covers cols [n16*16, +16), K [ks*32, +32).
//   elem index = (n16*64 + ks)*512 + lane*16 + e   (e in [0,16))
//   lane <16: col = n16*16+lane,     K = ks*32 +  0 + e
//   lane>=16: col = n16*16+lane-16,  K = ks*32 + 16 + e
// A wave's B-fragment load is then one contiguous 1 KB read (32 B/lane).
// ---------------------------------------------------------------------------
__global__ __launch_bounds__(256) void cvt_w_pack_kernel(
    const float* __restrict__ Wf, const float* __restrict__ Wi,
    const float* __restrict__ Wo, const float* __restrict__ Wc,
    bf16* __restrict__ out)
{
    size_t i  = (size_t)blockIdx.x * 256 + threadIdx.x;
    size_t e0 = i * 4;                       // 4 elems/thread, exact cover
    int within = (int)(e0 & 511);            // elem inside fragment
    size_t frag = e0 >> 9;
    int ks   = (int)(frag & 63);             // K step 0..63
    int n16  = (int)(frag >> 6);             // column group 0..255
    int lane = within >> 4;                  // 0..31
    int e    = within & 15;                  // 0,4,8,12 (i aligned to 4)

    int colg = n16 * 16 + (lane & 15);       // 0..4095 (row of W_all)
    int K    = ks * 32 + ((lane < 16) ? 0 : 16) + e;
    int gate = colg >> 10;
    int lrow = colg & 1023;
    const float* src =
        (gate == 0) ? Wf : (gate == 1) ? Wi : (gate == 2) ? Wo : Wc;
    f32x4 v = *(const f32x4*)(src + (size_t)lrow * DCAT + K);
    bf16* o = out + e0;
    o[0] = (bf16)v[0]; o[1] = (bf16)v[1]; o[2] = (bf16)v[2]; o[3] = (bf16)v[3];
}

__global__ __launch_bounds__(256) void cvt_x_kernel(
    const float* __restrict__ x, bf16* __restrict__ out)
{
    size_t e = ((size_t)blockIdx.x * 256 + threadIdx.x) * 4;
    f32x4 v = *(const f32x4*)(x + e);
    bf16* o = out + e;
    o[0] = (bf16)v[0]; o[1] = (bf16)v[1]; o[2] = (bf16)v[2]; o[3] = (bf16)v[3];
}

__global__ __launch_bounds__(256) void zero_kernel(unsigned int* __restrict__ p)
{
    p[(size_t)blockIdx.x * 256 + threadIdx.x] = 0u;
}

// ---------------------------------------------------------------------------
// One LSTM timestep.
//   grid = (32 n-slabs, 2 m-slabs), block = 256 threads (8 waves, wave32).
//   Block (nb, mb) computes gates[mb*32:+32, {g*1024 + nb*32 : +32}] for all 4
//   gates via bf16 WMMA, then does the gate math and updates c / h locally.
// ---------------------------------------------------------------------------
__global__ __launch_bounds__(256) void lstm_step_kernel(
    const bf16* __restrict__ Wpk,  // packed B fragments (16 MiB)
    const bf16* __restrict__ xb,   // [512, 64, 1024] bf16
    const bf16* __restrict__ hin,  // [64, 1024] bf16
    bf16*       __restrict__ hout, // [64, 1024] bf16
    float*      __restrict__ cbuf, // [64, 1024] f32
    const float* __restrict__ bF, const float* __restrict__ bI,
    const float* __restrict__ bO, const float* __restrict__ bC,
    float* __restrict__ out,       // [512, 64, 1024] f32
    int tstep)
{
    // LDS: double-buffered 32x64 bf16 z-chunk (stride 72 kills bank conflicts
    // on strided ds_load_b128 A-fragments) + 4-gate f32 staging tile.
    __shared__ bf16  sA[2][32 * 72];
    __shared__ float sG[4][32][33];

    const int tid  = threadIdx.x;
    const int lane = tid & 31;
    const int w    = tid >> 5;       // wave 0..7
    const int g    = w >> 1;         // gate 0..3 (f,i,o,c)
    const int tn   = w & 1;          // 16-col subtile within the 32-col slab
    const int n0   = blockIdx.x * 32;
    const int mb   = blockIdx.y;     // 0..1 (rows mb*32 .. +32)
    const int lhi  = (lane >= 16);
    const int l15  = lane & 15;

    // packed-B base for this wave: fragment group n16 = g*64 + nb*2 + tn,
    // per K-step ks the wave reads 1 KB contiguous (32 B per lane).
    const bf16* Bwave =
        Wpk + ((size_t)(g * 64 + blockIdx.x * 2 + tn) * 64) * 512 + lane * 16;

    v8f acc0 = {}; v8f acc1 = {};

    // cooperative chunk loader: 32 rows x 64 K, 8 bf16 (16 B) per thread
    const int lr   = tid >> 3;       // local row 0..31
    const int lsg  = tid & 7;        // 8-element segment 0..7
    const int grow = mb * 32 + lr;   // global batch row

    const bf16* xrow = xb + ((size_t)tstep * BATCH + grow) * DLAT;
    const bf16* hrow = hin + (size_t)grow * DLAT;

    u32x4 pf;
    {
        int kg = lsg * 8;            // chunk 0 is always in the h half
        pf = *(const u32x4*)(hrow + kg);
    }
    *(u32x4*)&sA[0][lr * 72 + lsg * 8] = pf;
    __syncthreads();

    const int NC = DCAT / 64;        // 32 chunks of K=64
    for (int c = 0; c < NC; ++c) {
        const int cur = c & 1;
        if (c + 1 < NC) {            // prefetch next chunk into registers
            int kg = (c + 1) * 64 + lsg * 8;
            const bf16* src = (kg < DLAT) ? (hrow + kg) : (xrow + (kg - DLAT));
            pf = *(const u32x4*)src;
        }
#pragma unroll
        for (int j = 0; j < 2; ++j) {
            Frag16 b;
            const bf16* bp = Bwave + (size_t)(c * 2 + j) * 512;
            b.q[0] = *(const u32x4*)bp;
            b.q[1] = *(const u32x4*)(bp + 8);

            const int kofs = j * 32 + (lhi ? 16 : 0);
            Frag16 a0, a1;
            a0.q[0] = *(const u32x4*)&sA[cur][(l15     ) * 72 + kofs];
            a0.q[1] = *(const u32x4*)&sA[cur][(l15     ) * 72 + kofs + 8];
            a1.q[0] = *(const u32x4*)&sA[cur][(16 + l15) * 72 + kofs];
            a1.q[1] = *(const u32x4*)&sA[cur][(16 + l15) * 72 + kofs + 8];

            acc0 = __builtin_amdgcn_wmma_f32_16x16x32_bf16(
                       false, a0.v, false, b.v, (short)0, acc0, false, false);
            acc1 = __builtin_amdgcn_wmma_f32_16x16x32_bf16(
                       false, a1.v, false, b.v, (short)0, acc1, false, false);
        }
        if (c + 1 < NC) {            // other buffer is free past the last sync
            *(u32x4*)&sA[(c + 1) & 1][lr * 72 + lsg * 8] = pf;
        }
        __syncthreads();
    }

    // spill accumulators: C/D layout is M = v + 8*(lane>=16), N = lane&15
#pragma unroll
    for (int v = 0; v < 8; ++v) {
        const int Ml = (lhi ? 8 : 0) + v;
        sG[g][Ml     ][tn * 16 + l15] = acc0[v];
        sG[g][16 + Ml][tn * 16 + l15] = acc1[v];
    }
    __syncthreads();

    // gate math + state update for this block's 32x32 region
#pragma unroll
    for (int i = 0; i < 4; ++i) {
        const int e   = tid + i * 256;   // 0..1023
        const int row = e >> 5;          // 0..31
        const int col = e & 31;
        const int ng  = n0 + col;

        const float fp = sG[0][row][col] + bF[ng];
        const float ip = sG[1][row][col] + bI[ng];
        const float op = sG[2][row][col] + bO[ng];
        const float gp = sG[3][row][col] + bC[ng];

        const float ft = 1.0f / (1.0f + expf(-fp));
        const float it = 1.0f / (1.0f + expf(-ip));
        const float ot = 1.0f / (1.0f + expf(-op));
        const float gt = tanhf(gp);

        const size_t ci = (size_t)(mb * 32 + row) * DLAT + ng;
        const float cn = ft * cbuf[ci] + it * gt;
        cbuf[ci] = cn;
        const float ht = ot * tanhf(cn);
        hout[ci] = (bf16)ht;
        out[(size_t)tstep * (BATCH * DLAT) + ci] = ht;
    }
}

// ---------------------------------------------------------------------------
// Host-side launch: convert once, then 512 dependent step kernels.
// Workspace layout (bytes):
//   [0,        16 MiB)   packed W bf16 (4096*2048*2)
//   [16 MiB,   80 MiB)   x bf16 (512*64*1024*2)
//   [+0,   +128 KiB)     h buffer 0 (bf16)
//   [+128, +256 KiB)     h buffer 1 (bf16)
//   [+256, +512 KiB)     c (f32)
// Total ~84.5 MB.
// ---------------------------------------------------------------------------
extern "C" void kernel_launch(void* const* d_in, const int* in_sizes, int n_in,
                              void* d_out, int out_size, void* d_ws, size_t ws_size,
                              hipStream_t stream)
{
    const float* x  = (const float*)d_in[0];
    const float* Wf = (const float*)d_in[1];
    const float* bF = (const float*)d_in[2];
    const float* Wi = (const float*)d_in[3];
    const float* bI = (const float*)d_in[4];
    const float* Wo = (const float*)d_in[5];
    const float* bO = (const float*)d_in[6];
    const float* Wc = (const float*)d_in[7];
    const float* bC = (const float*)d_in[8];
    float* out = (float*)d_out;

    char* ws = (char*)d_ws;
    bf16* Wpk = (bf16*)(ws);
    bf16* xb  = (bf16*)(ws + (size_t)16777216);
    bf16* h0  = (bf16*)(ws + (size_t)16777216 + 67108864);
    bf16* h1  = h0 + (size_t)BATCH * DLAT;
    float* cb = (float*)(ws + (size_t)16777216 + 67108864 + 262144);

    // pack weights / inputs to bf16 (exact element-count grids)
    cvt_w_pack_kernel<<<8192, 256, 0, stream>>>(Wf, Wi, Wo, Wc, Wpk);
    cvt_x_kernel<<<32768, 256, 0, stream>>>(x, xb);
    // zero h0, h1, c  (131072 dwords contiguous)
    zero_kernel<<<512, 256, 0, stream>>>((unsigned int*)h0);

    for (int t = 0; t < SEQ; ++t) {
        const bf16* hin = (t & 1) ? h1 : h0;
        bf16*      hout = (t & 1) ? h0 : h1;
        lstm_step_kernel<<<dim3(32, 2), 256, 0, stream>>>(
            Wpk, xb, hin, hout, cb, bF, bI, bO, bC, out, t);
    }
}
